// AttnBlock_3075196584598
// MI455X (gfx1250) — matrix operand order repaired and checked by reading, not compiled, not run on hardware
//
#include <hip/hip_runtime.h>
#include <math.h>
#include <stdint.h>

// ---------------------------------------------------------------------------
// AttnBlock for MI455X (gfx1250, wave32).  All matmuls on
// v_wmma_f32_16x16x32_f16 (f16 in / f32 accumulate).  Flash-attention style
// streaming (no 256MB score matrix).  K/V tiles staged into LDS with the
// CDNA5 async global->LDS path (ASYNCcnt) when the toolchain exposes it.
// ---------------------------------------------------------------------------

typedef __attribute__((ext_vector_type(16))) _Float16 v16h;
typedef __attribute__((ext_vector_type(8)))  _Float16 v8h;
typedef __attribute__((ext_vector_type(4)))  _Float16 v4h;
typedef __attribute__((ext_vector_type(8)))  float    v8f;
typedef __attribute__((ext_vector_type(4)))  int      v4i;

#define C_DIM 128
#define N_DIM 4096
#define B_DIM 4
#define NGRP  8
#define CPG   (C_DIM / NGRP)                 // 16 channels per group
#define QK_SCALE 0.08838834764831845f        // 128^-0.5

#define KT_STRIDE 144                        // halves; 288B = 9*32 -> 32B align
#define VT_STRIDE 80                         // halves; 160B = 5*32 -> 32B align
#define PT_STRIDE 80                         // halves
#define HT_STRIDE 144                        // halves

#if defined(__AMDGCN__) &&                                            \
    __has_builtin(__builtin_amdgcn_global_load_async_to_lds_b128) &&  \
    __has_builtin(__builtin_amdgcn_s_wait_asynccnt)
#define USE_ASYNC_LDS 1
#else
#define USE_ASYNC_LDS 0
#endif

// 16-byte global -> LDS copy.  Async (ASYNCcnt-tracked, no VGPR round trip)
// when available; plain VGPR-bounce copy otherwise.  The builtin is typed:
// (int4 AS1* src, int4 AS3* dst, imm offset, imm cpol).
static __device__ __forceinline__ void cp16_g2l(void* lds, const void* g) {
#if USE_ASYNC_LDS
  __builtin_amdgcn_global_load_async_to_lds_b128(
      (__attribute__((address_space(1))) v4i*)(unsigned long long)(size_t)g,
      (__attribute__((address_space(3))) v4i*)(unsigned)(size_t)lds,
      0, 0);
#else
  *(v8h*)lds = *(const v8h*)g;
#endif
}

static __device__ __forceinline__ void wait_async_lds() {
#if USE_ASYNC_LDS
  __builtin_amdgcn_s_wait_asynccnt(0);
#endif
}

static __device__ __forceinline__ v8f wmma_f16(v16h a, v16h b, v8f c) {
  // D = A(16x32 f16) * B(32x16 f16) + C(16x16 f32)
  return __builtin_amdgcn_wmma_f32_16x16x32_f16(false, a, false, b, (short)0, c,
                                                false, false);
}

static __device__ __forceinline__ v16h cat8(v8h lo, v8h hi) {
  v16h r;
#pragma unroll
  for (int i = 0; i < 8; ++i) { r[i] = lo[i]; r[8 + i] = hi[i]; }
  return r;
}

// ---------------------------------------------------------------------------
// 1) GroupNorm statistics: one workgroup per (batch, group), 16ch x 4096 elems
// ---------------------------------------------------------------------------
__global__ void gn_stats_kernel(const float* __restrict__ x,
                                float* __restrict__ stats) {
  const int bg = blockIdx.x;                           // 0..31
  const float4* p = (const float4*)(x + (size_t)bg * CPG * N_DIM);
  const int nq = CPG * N_DIM / 4;
  float s = 0.f, ss = 0.f;
  for (int i = threadIdx.x; i < nq; i += blockDim.x) {
    float4 v = p[i];
    s  += v.x + v.y + v.z + v.w;
    ss += v.x * v.x + v.y * v.y + v.z * v.z + v.w * v.w;
  }
  __shared__ float sh[512];
  sh[threadIdx.x] = s;
  sh[256 + threadIdx.x] = ss;
  __syncthreads();
  for (int o = 128; o > 0; o >>= 1) {
    if ((int)threadIdx.x < o) {
      sh[threadIdx.x] += sh[threadIdx.x + o];
      sh[256 + threadIdx.x] += sh[256 + threadIdx.x + o];
    }
    __syncthreads();
  }
  if (threadIdx.x == 0) {
    const float inv = 1.0f / (float)(CPG * N_DIM);
    const float mu  = sh[0] * inv;
    const float var = sh[256] * inv - mu * mu;
    stats[2 * bg]     = mu;
    stats[2 * bg + 1] = rsqrtf(var + 1e-6f);
  }
}

// ---------------------------------------------------------------------------
// 2) GroupNorm apply + f16 conversion, h stored (B, C, N)
// ---------------------------------------------------------------------------
__global__ void gn_apply_kernel(const float* __restrict__ x,
                                const float* __restrict__ gsc,
                                const float* __restrict__ gbi,
                                const float* __restrict__ stats,
                                _Float16* __restrict__ h) {
  const size_t qd   = (size_t)blockIdx.x * blockDim.x + threadIdx.x;
  const size_t base = qd * 4;
  const int c = (int)((base / N_DIM) % C_DIM);
  const int b = (int)(base / ((size_t)C_DIM * N_DIM));
  const int g = b * NGRP + c / CPG;
  const float mu = stats[2 * g], rsd = stats[2 * g + 1];
  const float sc = gsc[c] * rsd;
  const float bi = gbi[c] - mu * sc;
  float4 v = *(const float4*)(x + base);
  v4h o;
  o[0] = (_Float16)(v.x * sc + bi);
  o[1] = (_Float16)(v.y * sc + bi);
  o[2] = (_Float16)(v.z * sc + bi);
  o[3] = (_Float16)(v.w * sc + bi);
  *(v4h*)(h + base) = o;
}

// ---------------------------------------------------------------------------
// 3) Q/K/V projections.  grid = (N/16 tiles, 3 mats, B), block = 256 (8 waves)
//    Wave w computes out-channel tile [16w,16w+16) for one token tile.
//    q,k stored TOKEN-major (B,N,C); v stored CHANNEL-major (B,C,N).
//    q pre-scaled by C^-0.5.  h tile transposed into LDS once so B-fragments
//    are contiguous 32B reads.
// ---------------------------------------------------------------------------
__global__ void qkv_kernel(const _Float16* __restrict__ h,
                           const float* __restrict__ wq, const float* __restrict__ bq,
                           const float* __restrict__ wk, const float* __restrict__ bk,
                           const float* __restrict__ wv, const float* __restrict__ bv,
                           _Float16* __restrict__ qo, _Float16* __restrict__ ko,
                           _Float16* __restrict__ vo) {
  const int b = blockIdx.z, mat = blockIdx.y, n0 = blockIdx.x * 16;
  const float* w    = (mat == 0) ? wq : (mat == 1) ? wk : wv;
  const float* bias = (mat == 0) ? bq : (mat == 1) ? bk : bv;

  __shared__ _Float16 ht[16 * HT_STRIDE];              // [token][channel]
  const _Float16* hb = h + (size_t)b * C_DIM * N_DIM;
  {
    const int c = threadIdx.x >> 1, part = (threadIdx.x & 1) * 8;
    v8h t = *(const v8h*)&hb[(size_t)c * N_DIM + n0 + part];
#pragma unroll
    for (int j = 0; j < 8; ++j) ht[(part + j) * HT_STRIDE + c] = t[j];
  }
  __syncthreads();

  const int wave = threadIdx.x >> 5, lane = threadIdx.x & 31;
  const int m0 = wave * 16, m = lane & 15, hh = lane >> 4;

  // A fragments: weight tile (16 out-ch x 128 in-ch), f32 -> f16
  v16h A[4];
#pragma unroll
  for (int cc = 0; cc < 4; ++cc) {
    const float* wr = w + (size_t)(m0 + m) * C_DIM + cc * 32 + hh * 16;
    float lo[8], hi[8];
    *(float4*)&lo[0] = *(const float4*)wr;
    *(float4*)&lo[4] = *(const float4*)(wr + 4);
    *(float4*)&hi[0] = *(const float4*)(wr + 8);
    *(float4*)&hi[4] = *(const float4*)(wr + 12);
#pragma unroll
    for (int i = 0; i < 8; ++i) {
      A[cc][i]     = (_Float16)lo[i];
      A[cc][8 + i] = (_Float16)hi[i];
    }
  }

  // Preload all B fragments (one dscnt wait, no per-WMMA serialization)
  v16h Bf[4];
#pragma unroll
  for (int cc = 0; cc < 4; ++cc)
    Bf[cc] = *(const v16h*)&ht[m * HT_STRIDE + cc * 32 + hh * 16];

  v8f acc;
#pragma unroll
  for (int r = 0; r < 8; ++r) acc[r] = 0.f;
#pragma unroll
  for (int cc = 0; cc < 4; ++cc) acc = wmma_f16(A[cc], Bf[cc], acc);

  // C/D layout: reg r -> out-ch m0 + r + 8*hh, col -> token n0 + m
  if (mat != 2) {
    _Float16* out = (mat == 0) ? qo : ko;
    const float sc = (mat == 0) ? QK_SCALE : 1.0f;
    v8h pk;
#pragma unroll
    for (int r = 0; r < 8; ++r) {
      const int ch = m0 + 8 * hh + r;
      pk[r] = (_Float16)((acc[r] + bias[ch]) * sc);
    }
    *(v8h*)&out[((size_t)b * N_DIM + n0 + m) * C_DIM + m0 + 8 * hh] = pk;
  } else {
    // V channel-major (B, C, N)
#pragma unroll
    for (int r = 0; r < 8; ++r) {
      const int ch = m0 + 8 * hh + r;
      vo[((size_t)b * C_DIM + ch) * N_DIM + n0 + m] =
          (_Float16)(acc[r] + bias[ch]);
    }
  }
}

// ---------------------------------------------------------------------------
// 4) Flash attention.  grid = (N/64, B), block = 128 (4 waves, 16 queries ea).
//    Key tiles of 64 staged in LDS via async global->LDS copies; 32 WMMAs per
//    wave per tile; online softmax with 16-lane-half shuffle reductions.
// ---------------------------------------------------------------------------
__global__ void attn_kernel(const _Float16* __restrict__ q,
                            const _Float16* __restrict__ k,
                            const _Float16* __restrict__ v,
                            _Float16* __restrict__ ho) {
  const int b = blockIdx.y;
  const int wave = threadIdx.x >> 5, lane = threadIdx.x & 31;
  const int m = lane & 15, hh = lane >> 4;
  const int n0 = (blockIdx.x * 4 + wave) * 16;         // query base

  __shared__ _Float16 kt[64 * KT_STRIDE];              // [key][ch]
  __shared__ _Float16 vt[C_DIM * VT_STRIDE];           // [ch][key]
  __shared__ _Float16 pt[4][16 * PT_STRIDE];           // per-wave P [q][key]

  // Q fragments (queries pre-scaled by C^-0.5 in qkv_kernel)
  const _Float16* qb = q + ((size_t)b * N_DIM + n0) * C_DIM;
  v16h Aq[4];
#pragma unroll
  for (int cc = 0; cc < 4; ++cc) {
    const _Float16* r = qb + (size_t)m * C_DIM + cc * 32 + hh * 16;
    Aq[cc] = cat8(*(const v8h*)r, *(const v8h*)(r + 8));
  }

  v8f O[8];
#pragma unroll
  for (int f = 0; f < 8; ++f)
#pragma unroll
    for (int r = 0; r < 8; ++r) O[f][r] = 0.f;

  float rm[8], rs[8];
#pragma unroll
  for (int r = 0; r < 8; ++r) { rm[r] = -1e30f; rs[r] = 0.f; }

  const _Float16* kb = k + (size_t)b * N_DIM * C_DIM;              // (N,C)
  const _Float16* vb = v + (size_t)b * C_DIM * N_DIM;              // (C,N)

  for (int k0 = 0; k0 < N_DIM; k0 += 64) {
    __syncthreads();                                   // previous tile consumed
    // stage K tile (64 keys x 128 ch): 1024 16B chunks, 8 per thread
#pragma unroll
    for (int it = 0; it < 8; ++it) {
      const int chnk = threadIdx.x + it * 128;
      const int key = chnk >> 4, cg = (chnk & 15) * 8;
      cp16_g2l(&kt[key * KT_STRIDE + cg],
               &kb[(size_t)(k0 + key) * C_DIM + cg]);
    }
    // stage V tile (128 ch x 64 keys): channel-major source -> straight copy
#pragma unroll
    for (int it = 0; it < 8; ++it) {
      const int chnk = threadIdx.x + it * 128;
      const int c = chnk >> 3, ck = (chnk & 7) * 8;
      cp16_g2l(&vt[c * VT_STRIDE + ck],
               &vb[(size_t)c * N_DIM + k0 + ck]);
    }
    wait_async_lds();
    __syncthreads();

    // S = Q . K^T : four 16x16 f32 tiles (keys j*16..j*16+15)
    v8f S[4];
#pragma unroll
    for (int j = 0; j < 4; ++j)
#pragma unroll
      for (int r = 0; r < 8; ++r) S[j][r] = 0.f;
#pragma unroll
    for (int cc = 0; cc < 4; ++cc) {
#pragma unroll
      for (int j = 0; j < 4; ++j) {
        // B(k=ch, n=key): contiguous 16 halves at key row j*16+m
        v16h Bj = *(const v16h*)&kt[(j * 16 + m) * KT_STRIDE + cc * 32 + hh * 16];
        S[j] = wmma_f16(Aq[cc], Bj, S[j]);
      }
    }

    // --- online softmax: rows r+8*hh live in the lane's 16-lane half ---
    float mx[8], scl[8], ts[8];
#pragma unroll
    for (int r = 0; r < 8; ++r)
      mx[r] = fmaxf(fmaxf(S[0][r], S[1][r]), fmaxf(S[2][r], S[3][r]));
#pragma unroll
    for (int o = 1; o < 16; o <<= 1)
#pragma unroll
      for (int r = 0; r < 8; ++r) mx[r] = fmaxf(mx[r], __shfl_xor(mx[r], o));
#pragma unroll
    for (int r = 0; r < 8; ++r) {
      const float nm = fmaxf(rm[r], mx[r]);
      scl[r] = __expf(rm[r] - nm);
      rm[r]  = nm;
    }
#pragma unroll
    for (int r = 0; r < 8; ++r) {
      S[0][r] = __expf(S[0][r] - rm[r]);
      S[1][r] = __expf(S[1][r] - rm[r]);
      S[2][r] = __expf(S[2][r] - rm[r]);
      S[3][r] = __expf(S[3][r] - rm[r]);
      ts[r] = (S[0][r] + S[1][r]) + (S[2][r] + S[3][r]);
    }
#pragma unroll
    for (int o = 1; o < 16; o <<= 1)
#pragma unroll
      for (int r = 0; r < 8; ++r) ts[r] += __shfl_xor(ts[r], o);
#pragma unroll
    for (int r = 0; r < 8; ++r) rs[r] = rs[r] * scl[r] + ts[r];
#pragma unroll
    for (int f = 0; f < 8; ++f)
#pragma unroll
      for (int r = 0; r < 8; ++r) O[f][r] *= scl[r];

    // C/D layout -> A layout via per-wave LDS bounce
    _Float16* pw = &pt[wave][0];
#pragma unroll
    for (int j = 0; j < 4; ++j)
#pragma unroll
      for (int r = 0; r < 8; ++r)
        pw[(r + 8 * hh) * PT_STRIDE + j * 16 + m] = (_Float16)S[j][r];
    __syncthreads();

    // O += P . V  (two 32-key chunks x 8 channel tiles)
#pragma unroll
    for (int kk = 0; kk < 2; ++kk) {
      const _Float16* pr = &pt[wave][m * PT_STRIDE + kk * 32 + hh * 16];
      v16h Ap = cat8(*(const v8h*)pr, *(const v8h*)(pr + 8));
#pragma unroll
      for (int f = 0; f < 8; ++f) {
        // B(k=key, n=ch): contiguous 16 halves in vt row ch=f*16+m
        v16h Bv =
            *(const v16h*)&vt[(f * 16 + m) * VT_STRIDE + kk * 32 + hh * 16];
        O[f] = wmma_f16(Ap, Bv, O[f]);
      }
    }
  }

#pragma unroll
  for (int r = 0; r < 8; ++r) rs[r] = 1.0f / rs[r];
  _Float16* ob = ho + ((size_t)b * N_DIM + n0) * C_DIM;   // token-major out
#pragma unroll
  for (int f = 0; f < 8; ++f)
#pragma unroll
    for (int r = 0; r < 8; ++r) {
      const int row = r + 8 * hh;
      ob[(size_t)row * C_DIM + f * 16 + m] = (_Float16)(O[f][r] * rs[r]);
    }
}

// ---------------------------------------------------------------------------
// 5) Output projection + bias + residual.  grid = (N/16, B), block = 256.
// ---------------------------------------------------------------------------
__global__ void proj_kernel(const _Float16* __restrict__ ho,
                            const float* __restrict__ wp,
                            const float* __restrict__ bp,
                            const float* __restrict__ x,
                            float* __restrict__ out) {
  const int b = blockIdx.y, n0 = blockIdx.x * 16;
  const int wave = threadIdx.x >> 5, lane = threadIdx.x & 31;
  const int m0 = wave * 16, m = lane & 15, hh = lane >> 4;

  v16h A[4];
#pragma unroll
  for (int cc = 0; cc < 4; ++cc) {
    const float* wr = wp + (size_t)(m0 + m) * C_DIM + cc * 32 + hh * 16;
    float lo[8], hi[8];
    *(float4*)&lo[0] = *(const float4*)wr;
    *(float4*)&lo[4] = *(const float4*)(wr + 4);
    *(float4*)&hi[0] = *(const float4*)(wr + 8);
    *(float4*)&hi[4] = *(const float4*)(wr + 12);
#pragma unroll
    for (int i = 0; i < 8; ++i) {
      A[cc][i]     = (_Float16)lo[i];
      A[cc][8 + i] = (_Float16)hi[i];
    }
  }

  const _Float16* hb = ho + ((size_t)b * N_DIM + n0 + m) * C_DIM;
  v8f acc;
#pragma unroll
  for (int r = 0; r < 8; ++r) acc[r] = 0.f;
#pragma unroll
  for (int cc = 0; cc < 4; ++cc) {
    // token-major: 16 contiguous halves, 32B aligned
    v16h Bf = *(const v16h*)(hb + cc * 32 + hh * 16);
    acc = wmma_f16(A[cc], Bf, acc);
  }

#pragma unroll
  for (int r = 0; r < 8; ++r) {
    const int c = m0 + r + 8 * hh;
    const size_t idx = ((size_t)b * C_DIM + c) * N_DIM + n0 + m;
    out[idx] = acc[r] + bp[c] + x[idx];
  }
}

// ---------------------------------------------------------------------------
extern "C" void kernel_launch(void* const* d_in, const int* in_sizes, int n_in,
                              void* d_out, int out_size, void* d_ws,
                              size_t ws_size, hipStream_t stream) {
  (void)in_sizes; (void)n_in; (void)out_size; (void)ws_size;
  const float* x   = (const float*)d_in[0];
  const float* gsc = (const float*)d_in[1];
  const float* gbi = (const float*)d_in[2];
  const float* wq  = (const float*)d_in[3];
  const float* bq  = (const float*)d_in[4];
  const float* wk  = (const float*)d_in[5];
  const float* bk  = (const float*)d_in[6];
  const float* wv  = (const float*)d_in[7];
  const float* bv  = (const float*)d_in[8];
  const float* wp  = (const float*)d_in[9];
  const float* bp  = (const float*)d_in[10];

  char* ws = (char*)d_ws;
  const size_t ELEMS = (size_t)B_DIM * C_DIM * N_DIM;  // 2,097,152
  float*    stats = (float*)ws;                        // 256 B used
  _Float16* h16   = (_Float16*)(ws + 1024);            // reused as attn output
  _Float16* q16   = h16 + ELEMS;
  _Float16* k16   = q16 + ELEMS;
  _Float16* v16   = k16 + ELEMS;                       // total ~16.8 MB

  gn_stats_kernel<<<dim3(B_DIM * NGRP), dim3(256), 0, stream>>>(x, stats);
  gn_apply_kernel<<<dim3((unsigned)(ELEMS / 4 / 256)), dim3(256), 0, stream>>>(
      x, gsc, gbi, stats, h16);
  qkv_kernel<<<dim3(N_DIM / 16, 3, B_DIM), dim3(256), 0, stream>>>(
      h16, wq, bq, wk, bk, wv, bv, q16, k16, v16);
  attn_kernel<<<dim3(N_DIM / 64, B_DIM), dim3(128), 0, stream>>>(
      q16, k16, v16, h16 /* ho aliases h16 */);
  proj_kernel<<<dim3(N_DIM / 16, B_DIM), dim3(256), 0, stream>>>(
      h16, wp, bp, x, (float*)d_out);
}
